// MultiHeadAttentionAwareTemporalContex_q1d_k1d_17489106829906
// MI455X (gfx1250) — compile-verified, matrix operand order checked by reading, not run on hardware
//
#include <hip/hip_runtime.h>

typedef __attribute__((ext_vector_type(16))) _Float16 v16h;
typedef __attribute__((ext_vector_type(8)))  float    v8f;

#define D_  512
#define T_  24
#define H_  8
#define DK_ 64
#define SLAB 36864              // one 32x512 weight slab, padded: 512 * 36 halves * 2B
#define XBUF_B (34 * D_ * 2)    // proj xbuf bytes
#define WOOFF 147456            // attn: phase-2 Wo slab / phase-1 score overlay offset

union FragU { v16h h; uint2 u2[4]; _Float16 e[16]; };

// Load a 16-half A/B WMMA fragment whose 16 reduction-dim elements are
// CONTIGUOUS in LDS starting at p. Element pattern per ISA 16-bit layout:
// lane<16 : K = {0..7, 16..23}; lane>=16 : K = {8..15, 24..31}  (hi = lane>>4)
__device__ __forceinline__ v16h ldfrag(const _Float16* p, int hi) {
  FragU f;
  const uint2* a = (const uint2*)(p + hi * 8);
  const uint2* b = (const uint2*)(p + 16 + hi * 8);
  f.u2[0] = a[0]; f.u2[1] = a[1];
  f.u2[2] = b[0]; f.u2[3] = b[1];
  return f.h;
}

// Same pattern, reduction elements strided in LDS (stride in halves).
__device__ __forceinline__ v16h ldfrag_strided(const _Float16* p, int hi, int stride) {
  FragU f;
#pragma unroll
  for (int e = 0; e < 16; ++e) {
    int off = ((e < 8) ? e : (e + 8)) + hi * 8;
    f.e[e] = p[off * stride];
  }
  return f.h;
}

__device__ __forceinline__ v8f wmma16(v16h a, v16h b, v8f c) {
  return __builtin_amdgcn_wmma_f32_16x16x32_f16(false, a, false, b, (short)0, c,
                                                false, false);
}

// ---- gfx1250 async global->LDS copy (ASYNCcnt-tracked), 16B per lane ----
__device__ __forceinline__ void async_b128(unsigned lds_off, const void* g) {
  asm volatile("global_load_async_to_lds_b128 %0, %1, off"
               :: "v"(lds_off), "v"(g) : "memory");
}
__device__ __forceinline__ void wait_async9() {
  asm volatile("s_wait_asynccnt 9" ::: "memory");
}
__device__ __forceinline__ void wait_async0() {
  asm volatile("s_wait_asynccnt 0" ::: "memory");
}

// stage one padded weight slab (36864B) into LDS at lds_off, 9 async B128/lane
__device__ __forceinline__ void stage_slab(const void* wbase, int rs,
                                           unsigned lds_off, int tid) {
  const char* src = (const char*)wbase + (size_t)rs * SLAB + tid * 16;
  unsigned dst = lds_off + tid * 16;
#pragma unroll
  for (int it = 0; it < SLAB / 4096; ++it)
    async_b128(dst + it * 4096, src + it * 4096);
}

// ---- weight prep: conv W (O,I,1,K) f32 -> slabbed/transposed/padded f16 ----
// reduction row r = kk*512 + i ; slab rs = r>>5, rr = r&31
// WBt[rs*512*36 + o*36 + rr] = W[o][i][kk]
__global__ __launch_bounds__(256) void prep_conv_w(const float* __restrict__ W,
                                                   _Float16* __restrict__ out,
                                                   int taps) {
  int j = blockIdx.x * 256 + threadIdx.x;
  if (j >= taps * D_ * D_) return;
  int o  = j & (D_ - 1);
  int r  = j >> 9;
  int i  = r & (D_ - 1);
  int kk = r >> 9;
  out[((size_t)(r >> 5) * D_ + o) * 36 + (r & 31)] =
      (_Float16)W[(o * D_ + i) * taps + kk];
}

// ---- weight prep: linear W (E,D) f32 -> slabbed B[d][e] = W[e][d] ----
__global__ __launch_bounds__(256) void prep_lin_w(const float* __restrict__ W,
                                                  _Float16* __restrict__ out) {
  int j = blockIdx.x * 256 + threadIdx.x;
  if (j >= D_ * D_) return;
  int e = j & (D_ - 1);
  int d = j >> 9;
  out[((size_t)(d >> 5) * D_ + e) * 36 + (d & 31)] = (_Float16)W[e * D_ + d];
}

// ---- conv / linear projection as WMMA GEMM, one block per (b,n) ----
__global__ __launch_bounds__(256) void proj_kernel(const float* __restrict__ X,
                                                   const _Float16* __restrict__ WBt,
                                                   const float* __restrict__ bias,
                                                   _Float16* __restrict__ out,
                                                   int taps, int pad) {
  extern __shared__ char smem[];
  _Float16* xbuf = (_Float16*)smem;                   // 34 x 512 halves
  const unsigned lbase = __builtin_amdgcn_groupstaticsize();

  const int tid = threadIdx.x;
  const int lane = tid & 31, wave = tid >> 5;
  const int hi = lane >> 4, ln = lane & 15;
  const long long bn = blockIdx.x;
  const float* Xb   = X   + bn * (long long)(T_ * D_);
  _Float16*    outb = out + bn * (long long)(T_ * D_);

  const uint4 z4 = {0, 0, 0, 0};
  for (int i = tid; i < (34 * D_) / 8; i += 256) ((uint4*)xbuf)[i] = z4;
  __syncthreads();
  for (int i = tid; i < T_ * D_; i += 256)            // rows 1..24 hold t=0..23
    xbuf[((i >> 9) + 1) * D_ + (i & 511)] = (_Float16)Xb[i];

  v8f zero = {};
  v8f acc[2][4];
#pragma unroll
  for (int m = 0; m < 2; ++m)
#pragma unroll
    for (int n = 0; n < 4; ++n) acc[m][n] = zero;

  const int NR = taps * 16;
  const int n0 = wave * 4;
  stage_slab(WBt, 0, lbase + XBUF_B, tid);
  for (int rs = 0; rs < NR; ++rs) {
    if (rs + 1 < NR) {
      stage_slab(WBt, rs + 1, lbase + XBUF_B + ((rs + 1) & 1) * SLAB, tid);
      wait_async9();
    } else {
      wait_async0();
    }
    __syncthreads();                                  // slab rs ready everywhere
    const _Float16* bslab = (const _Float16*)(smem + XBUF_B + (rs & 1) * SLAB);
    int kk = rs >> 4;
    int i0 = (rs & 15) * 32;
    v16h a[2];
#pragma unroll
    for (int m = 0; m < 2; ++m) {
      int xrow = m * 16 + ln + kk - pad + 1;          // conv shift, halo rows zero
      a[m] = ldfrag(&xbuf[xrow * D_ + i0], hi);
    }
#pragma unroll
    for (int n = 0; n < 4; ++n) {
      int o = (n0 + n) * 16 + ln;
      v16h b = ldfrag(&bslab[o * 36], hi);
#pragma unroll
      for (int m = 0; m < 2; ++m) acc[m][n] = wmma16(a[m], b, acc[m][n]);
    }
    __syncthreads();                                  // before slab reuse
  }

#pragma unroll
  for (int m = 0; m < 2; ++m)
#pragma unroll
    for (int n = 0; n < 4; ++n) {
      int o = (n0 + n) * 16 + ln;
      float bb = bias[o];
#pragma unroll
      for (int v = 0; v < 8; ++v) {
        int t = m * 16 + hi * 8 + v;
        if (t < T_) outb[t * D_ + o] = (_Float16)(acc[m][n][v] + bb);
      }
    }
}

// ---- fused attention (one wave per head) + output projection ----
__global__ __launch_bounds__(256) void attn_kernel(const _Float16* __restrict__ qH,
                                                   const _Float16* __restrict__ kH,
                                                   const _Float16* __restrict__ vH,
                                                   const _Float16* __restrict__ WoBt,
                                                   const float* __restrict__ bo,
                                                   float* __restrict__ out) {
  extern __shared__ char smem[];
  _Float16* qb = (_Float16*)smem;                     // 32 x 512
  _Float16* kb = (_Float16*)(smem + 32768);
  _Float16* vb = (_Float16*)(smem + 65536);
  _Float16* xb = (_Float16*)(smem + 98304);
  _Float16* pb = (_Float16*)(smem + 131072);          // 8 x 32 x 32 probs
  float*    sb = (float*)(smem + WOOFF);              // 8 x 24 x 32 scores (phase 1)
  const unsigned lbase = __builtin_amdgcn_groupstaticsize();

  const int tid = threadIdx.x;
  const int lane = tid & 31, wave = tid >> 5;
  const int hi = lane >> 4, ln = lane & 15;
  const long long bn   = blockIdx.x;
  const long long base = bn * (long long)(T_ * D_);

  // async-load q/k/v tiles (24x512 f16 = 24576B each, 6 B128 per lane each)
#pragma unroll
  for (int it = 0; it < 6; ++it) {
    unsigned off = tid * 16 + it * 4096;
    async_b128(lbase + 0u     + off, (const char*)(qH + base) + off);
    async_b128(lbase + 32768u + off, (const char*)(kH + base) + off);
    async_b128(lbase + 65536u + off, (const char*)(vH + base) + off);
  }
  const uint4 z4 = {0, 0, 0, 0};
  for (int i = tid; i < (8 * D_) / 8; i += 256) {     // zero pad rows 24..31
    ((uint4*)(qb + 24 * D_))[i] = z4;
    ((uint4*)(kb + 24 * D_))[i] = z4;
    ((uint4*)(vb + 24 * D_))[i] = z4;
  }
  for (int i = tid; i < (32 * D_) / 8; i += 256) ((uint4*)xb)[i] = z4;
  for (int i = tid; i < (8 * 32 * 32) / 8; i += 256) ((uint4*)pb)[i] = z4;
  wait_async0();
  __syncthreads();

  const int h = wave;  // one head per wave
  v8f zero = {};

  // scores = (q_h . k_h^T) / 8 : DK=64 -> 2 WMMA k-steps
  v8f sc[2][2];
#pragma unroll
  for (int m = 0; m < 2; ++m)
#pragma unroll
    for (int n = 0; n < 2; ++n) sc[m][n] = zero;
#pragma unroll
  for (int ks = 0; ks < 2; ++ks) {
    int dbase = h * DK_ + ks * 32;
    v16h aq[2];
#pragma unroll
    for (int m = 0; m < 2; ++m)
      aq[m] = ldfrag(&qb[(m * 16 + ln) * D_ + dbase], hi);
#pragma unroll
    for (int n = 0; n < 2; ++n) {
      int s = n * 16 + ln;
      v16h bk = ldfrag(&kb[s * D_ + dbase], hi);      // B[d][s] = k[s][d]
#pragma unroll
      for (int m = 0; m < 2; ++m) sc[m][n] = wmma16(aq[m], bk, sc[m][n]);
    }
  }
#pragma unroll
  for (int m = 0; m < 2; ++m)
#pragma unroll
    for (int n = 0; n < 2; ++n) {
      int s = n * 16 + ln;
#pragma unroll
      for (int v = 0; v < 8; ++v) {
        int t = m * 16 + hi * 8 + v;
        if (t < T_ && s < T_) sb[(h * T_ + t) * 32 + s] = sc[m][n][v] * 0.125f;
      }
    }
  __syncthreads();

  // causal softmax, one row per lane
  if (lane < T_) {
    int t = lane;
    float* row = &sb[(h * T_ + t) * 32];
    float mx = -1e30f;
    for (int s = 0; s <= t; ++s) mx = fmaxf(mx, row[s]);
    float sum = 0.f;
    for (int s = 0; s <= t; ++s) { float ex = __expf(row[s] - mx); row[s] = ex; sum += ex; }
    float inv = 1.f / sum;
    _Float16* prow = &pb[(h * 32 + t) * 32];
    for (int s = 0; s <= t; ++s) prow[s] = (_Float16)(row[s] * inv);
  }
  __syncthreads();

  // x_h = p_h . v_h : N=64 (4 tiles), K padded to 32 -> 1 WMMA step
  v8f xc[2][4];
#pragma unroll
  for (int m = 0; m < 2; ++m)
#pragma unroll
    for (int n = 0; n < 4; ++n) xc[m][n] = zero;
  v16h ap[2];
#pragma unroll
  for (int m = 0; m < 2; ++m)
    ap[m] = ldfrag(&pb[(h * 32 + m * 16 + ln) * 32], hi);
#pragma unroll
  for (int n = 0; n < 4; ++n) {
    int dcol = h * DK_ + n * 16 + ln;
    v16h bv = ldfrag_strided(&vb[dcol], hi, D_);      // B[s][d] = v[s][dcol]
#pragma unroll
    for (int m = 0; m < 2; ++m) xc[m][n] = wmma16(ap[m], bv, xc[m][n]);
  }
#pragma unroll
  for (int m = 0; m < 2; ++m)
#pragma unroll
    for (int n = 0; n < 4; ++n) {
      int c = h * DK_ + n * 16 + ln;
#pragma unroll
      for (int v = 0; v < 8; ++v) {
        int t = m * 16 + hi * 8 + v;
        if (t < T_) xb[t * D_ + c] = (_Float16)xc[m][n][v];
      }
    }
  __syncthreads();                                    // sb dead, xb complete

  // out = x . Wo^T + bo : 16 k-steps, double-buffered async Wo slabs
  v8f acc[2][4];
#pragma unroll
  for (int m = 0; m < 2; ++m)
#pragma unroll
    for (int n = 0; n < 4; ++n) acc[m][n] = zero;
  const int n0 = wave * 4;
  stage_slab(WoBt, 0, lbase + WOOFF, tid);
  for (int rs = 0; rs < 16; ++rs) {
    if (rs + 1 < 16) {
      stage_slab(WoBt, rs + 1, lbase + WOOFF + ((rs + 1) & 1) * SLAB, tid);
      wait_async9();
    } else {
      wait_async0();
    }
    __syncthreads();
    const _Float16* wslab = (const _Float16*)(smem + WOOFF + (rs & 1) * SLAB);
    int i0 = rs * 32;
    v16h a[2];
#pragma unroll
    for (int m = 0; m < 2; ++m)
      a[m] = ldfrag(&xb[(m * 16 + ln) * D_ + i0], hi);
#pragma unroll
    for (int n = 0; n < 4; ++n) {
      int o = (n0 + n) * 16 + ln;
      v16h b = ldfrag(&wslab[o * 36], hi);
#pragma unroll
      for (int m = 0; m < 2; ++m) acc[m][n] = wmma16(a[m], b, acc[m][n]);
    }
    __syncthreads();
  }
  float* outb = out + base;
#pragma unroll
  for (int m = 0; m < 2; ++m)
#pragma unroll
    for (int n = 0; n < 4; ++n) {
      int o = (n0 + n) * 16 + ln;
      float bb = bo[o];
#pragma unroll
      for (int v = 0; v < 8; ++v) {
        int t = m * 16 + hi * 8 + v;
        if (t < T_) outb[t * D_ + o] = acc[m][n][v] + bb;
      }
    }
}

extern "C" void kernel_launch(void* const* d_in, const int* in_sizes, int n_in,
                              void* d_out, int out_size, void* d_ws, size_t ws_size,
                              hipStream_t stream) {
  (void)in_sizes; (void)n_in; (void)out_size; (void)ws_size;
  const float* query = (const float*)d_in[0];
  const float* key_  = (const float*)d_in[1];
  const float* value = (const float*)d_in[2];
  // d_in[3] = mask (tril causal) -- enforced analytically in the kernel
  const float* Wq = (const float*)d_in[4];
  const float* bq = (const float*)d_in[5];
  const float* Wk = (const float*)d_in[6];
  const float* bk = (const float*)d_in[7];
  const float* Wv = (const float*)d_in[8];
  const float* bv = (const float*)d_in[9];
  const float* Wo = (const float*)d_in[10];
  const float* bo = (const float*)d_in[11];
  float* out = (float*)d_out;

  char* ws = (char*)d_ws;
  const size_t CONVW = 48ULL * SLAB;                  // 1769472 B
  const size_t LINW  = 16ULL * SLAB;                  //  589824 B
  const long long QKV = 8LL * 512 * T_ * D_ * 2;      // 100663296 B each
  _Float16* WqBt = (_Float16*)(ws);
  _Float16* WkBt = (_Float16*)(ws + CONVW);
  _Float16* WvBt = (_Float16*)(ws + 2 * CONVW);
  _Float16* WoBt = (_Float16*)(ws + 2 * CONVW + LINW);
  char*     actb = ws + 2 * CONVW + 2 * LINW;
  _Float16* qHt  = (_Float16*)(actb);
  _Float16* kHt  = (_Float16*)(actb + QKV);
  _Float16* vHt  = (_Float16*)(actb + 2 * QKV);

  const int BN = 8 * 512;
  prep_conv_w<<<(3 * D_ * D_) / 256, 256, 0, stream>>>(Wq, WqBt, 3);
  prep_conv_w<<<(3 * D_ * D_) / 256, 256, 0, stream>>>(Wk, WkBt, 3);
  prep_lin_w<<<(D_ * D_) / 256, 256, 0, stream>>>(Wv, WvBt);
  prep_lin_w<<<(D_ * D_) / 256, 256, 0, stream>>>(Wo, WoBt);

  size_t smem2 = XBUF_B + 2 * SLAB;                   // 108544 B
  proj_kernel<<<BN, 256, smem2, stream>>>(query, WqBt, bq, qHt, 3, 1);
  proj_kernel<<<BN, 256, smem2, stream>>>(key_,  WkBt, bk, kHt, 3, 1);
  proj_kernel<<<BN, 256, smem2, stream>>>(value, WvBt, bv, vHt, 1, 0);

  size_t smem3 = WOOFF + 2 * SLAB;                    // 221184 B
  attn_kernel<<<BN, 256, smem3, stream>>>(qHt, kHt, vHt, WoBt, bo, out);
}
